// LigerFusedLinearAIXueLoss_72224170049607
// MI455X (gfx1250) — compile-verified, matrix operand order checked by reading, not run on hardware
//
#include <hip/hip_runtime.h>

// ---------------------------------------------------------------------------
// GRPO fused linear loss for MI455X (gfx1250, wave32, WMMA).
// Phase A: one-pass fp32 -> bf16 hi/lo split of X and W into workspace.
// Phase B: logits GEMM (4096x32000x2048) fused with online logsumexp,
//          3x v_wmma_f32_16x16x32_bf16 per K-step (hi*hi + hi*lo + lo*hi),
//          WMMAs interleaved across 4 accumulators; high register budget so
//          all fragments stay live (no reload stalls / WAR v_nops).
// Phase C: exact fp32 selected-token logit; Phase D: loss/metrics reduction.
// ---------------------------------------------------------------------------

typedef __attribute__((ext_vector_type(16))) __bf16 v16bf;
typedef __attribute__((ext_vector_type(8)))  __bf16 v8bf;
typedef __attribute__((ext_vector_type(4)))  __bf16 v4bf;
typedef __attribute__((ext_vector_type(8)))  float  v8f;

union Frag16 { v16bf v; v8bf h[2]; };

#define B_ 4
#define T_ 1024
#define H_ 2048
#define V_ 32000
#define M_ (B_ * T_)          // 4096 token rows
#define NCHUNK 10             // V split across grid.y
#define CHUNK_N (V_ / NCHUNK) // 3200 columns per chunk
#define NSLOTS (NCHUNK * 2)   // 2 N-waves per WG each own a slot

// ---------------------------------------------------------------------------
// Phase A: split fp32 tensor into hi/lo bf16 (x = hi + lo), 4 elems/thread.
// ---------------------------------------------------------------------------
__global__ __launch_bounds__(256)
void split_bf16(const float* __restrict__ src, __bf16* __restrict__ hi,
                __bf16* __restrict__ lo, int n4)
{
  const int i = blockIdx.x * 256 + threadIdx.x;
  if (i >= n4) return;
  const float4 v = ((const float4*)src)[i];
  const float vv[4] = {v.x, v.y, v.z, v.w};
  v4bf hv, lv;
#pragma unroll
  for (int j = 0; j < 4; ++j) {
    __bf16 h = (__bf16)vv[j];
    hv[j] = h;
    lv[j] = (__bf16)(vv[j] - (float)h);
  }
  ((v4bf*)hi)[i] = hv;
  ((v4bf*)lo)[i] = lv;
}

// ---------------------------------------------------------------------------
// Phase B: tiled GEMM with fused online softmax statistics.
// Block: 256 threads = 8 waves, arranged 4 (M) x 2 (N).
// Each wave: 16 rows x 64 cols per pass (4 WMMA C tiles), K loop over H.
// Per-lane online (max,sumexp); single cross-lane combine in the epilogue.
// launch_bounds(..., 1): allow a large VGPR budget (2 waves/SIMD) so the
// whole K-step load clause stays live in registers.
// ---------------------------------------------------------------------------
__global__ __launch_bounds__(256, 1)
void grpo_lse_gemm(const __bf16* __restrict__ Xhi, const __bf16* __restrict__ Xlo,
                   const __bf16* __restrict__ Whi, const __bf16* __restrict__ Wlo,
                   const float* __restrict__ bias,
                   float* __restrict__ pmax, float* __restrict__ psum)
{
  const int lane = threadIdx.x & 31;
  const int wave = threadIdx.x >> 5;
  const int wm   = wave >> 1;       // 0..3  (M group)
  const int wn   = wave & 1;        // 0..1  (N split)
  const int half = lane >> 4;       // lane half selects K stripe
  const int l15  = lane & 15;
  const int mbase = blockIdx.x * 64 + wm * 16;
  const int chunk = blockIdx.y;
  const int c0 = half * 8;          // K offset for this lane half

  const __bf16* xh = Xhi + (size_t)(mbase + l15) * H_ + c0;
  const __bf16* xl = Xlo + (size_t)(mbase + l15) * H_ + c0;

  float rmax[8], rsum[8];           // lane-local running stats per C row
#pragma unroll
  for (int i = 0; i < 8; ++i) { rmax[i] = -__builtin_inff(); rsum[i] = 0.f; }

  for (int p = 0; p < CHUNK_N / 128; ++p) {
    const int nbase = chunk * CHUNK_N + p * 128 + wn * 64;
    // Single base pointer per stream; per-tile row offset t*16*H_ (<= 192KB)
    // folds into the load instruction's 24-bit immediate offset.
    const __bf16* whb = Whi + (size_t)(nbase + l15) * H_ + c0;
    const __bf16* wlb = Wlo + (size_t)(nbase + l15) * H_ + c0;
    v8f acc4[4] = {};

#pragma unroll 1
    for (int k0 = 0; k0 < H_; k0 += 32) {
      // Load ALL fragments first: one deep load clause per K-step.
      Frag16 ah, al, bh[4], bl[4];
      ah.h[0] = *(const v8bf*)(xh + k0);
      ah.h[1] = *(const v8bf*)(xh + k0 + 16);
      al.h[0] = *(const v8bf*)(xl + k0);
      al.h[1] = *(const v8bf*)(xl + k0 + 16);
#pragma unroll
      for (int t = 0; t < 4; ++t) {
        const size_t ro = (size_t)t * 16 * H_;
        bh[t].h[0] = *(const v8bf*)(whb + ro + k0);
        bh[t].h[1] = *(const v8bf*)(whb + ro + k0 + 16);
        bl[t].h[0] = *(const v8bf*)(wlb + ro + k0);
        bl[t].h[1] = *(const v8bf*)(wlb + ro + k0 + 16);
      }
      __builtin_prefetch(xh + k0 + 512, 0, 0);  // global_prefetch_b8

      // Interleave across accumulators: dependent WMMAs on the same acc are
      // 4 apart, covering the bf16 WMMA->WMMA hazard window without v_nops.
#pragma unroll
      for (int t = 0; t < 4; ++t)
        acc4[t] = __builtin_amdgcn_wmma_f32_16x16x32_bf16(
            false, ah.v, false, bh[t].v, (short)0, acc4[t], false, false);
#pragma unroll
      for (int t = 0; t < 4; ++t)
        acc4[t] = __builtin_amdgcn_wmma_f32_16x16x32_bf16(
            false, ah.v, false, bl[t].v, (short)0, acc4[t], false, false);
#pragma unroll
      for (int t = 0; t < 4; ++t)
        acc4[t] = __builtin_amdgcn_wmma_f32_16x16x32_bf16(
            false, al.v, false, bh[t].v, (short)0, acc4[t], false, false);
    }

    // Lane-local online softmax update (no cross-lane traffic in the loop).
#pragma unroll
    for (int t = 0; t < 4; ++t) {
      const float bv = bias[nbase + t * 16 + l15];
#pragma unroll
      for (int i = 0; i < 8; ++i) {
        // C layout: VGPR i, lanes 0-15 -> row i; lanes 16-31 -> row i+8.
        const float v  = acc4[t][i] + bv;
        const float nm = fmaxf(rmax[i], v);
        rsum[i] = rsum[i] * __expf(rmax[i] - nm) + __expf(v - nm);
        rmax[i] = nm;
      }
    }
  }

  // Epilogue: combine (max,sum) pairs across the 16-lane half (4 butterflies).
#pragma unroll
  for (int i = 0; i < 8; ++i) {
#pragma unroll
    for (int m = 1; m <= 8; m <<= 1) {
      const float om = __shfl_xor(rmax[i], m);
      const float os = __shfl_xor(rsum[i], m);
      const float nm = fmaxf(rmax[i], om);
      rsum[i] = rsum[i] * __expf(rmax[i] - nm) + os * __expf(om - nm);
      rmax[i] = nm;
    }
  }

  if (l15 == 0) {
    const int rbase = mbase + half * 8;
    const int slot = chunk * 2 + wn;
#pragma unroll
    for (int i = 0; i < 8; ++i) {
      pmax[(size_t)(rbase + i) * NSLOTS + slot] = rmax[i];
      psum[(size_t)(rbase + i) * NSLOTS + slot] = rsum[i];
    }
  }
}

// ---------------------------------------------------------------------------
// Phase C: exact fp32 selected-token logit (one wave per row).
// ---------------------------------------------------------------------------
__global__ __launch_bounds__(256)
void tok_logit_kernel(const float* __restrict__ X, const float* __restrict__ W,
                      const float* __restrict__ bias,
                      const int* __restrict__ tok, float* __restrict__ tokl)
{
  const int wid  = (int)((blockIdx.x * blockDim.x + threadIdx.x) >> 5);
  const int lane = threadIdx.x & 31;
  if (wid >= M_) return;
  const int t = tok[wid];
  const float* x = X + (size_t)wid * H_;
  const float* w = W + (size_t)t * H_;
  float s = 0.f;
#pragma unroll
  for (int j = 0; j < H_ / 128; ++j) {
    float4 xv = *(const float4*)(x + j * 128 + lane * 4);
    float4 wv = *(const float4*)(w + j * 128 + lane * 4);
    s += xv.x * wv.x + xv.y * wv.y + xv.z * wv.z + xv.w * wv.w;
  }
  s += __shfl_xor(s, 1);
  s += __shfl_xor(s, 2);
  s += __shfl_xor(s, 4);
  s += __shfl_xor(s, 8);
  s += __shfl_xor(s, 16);
  if (lane == 0) tokl[wid] = s + bias[t];
}

// ---------------------------------------------------------------------------
// Phase D: combine partials -> lse -> per-token loss; block reduce + atomics.
// ---------------------------------------------------------------------------
__device__ inline float block_sum256(float v, float* sm) {
  const int tid = threadIdx.x;
  sm[tid] = v;
  __syncthreads();
  for (int s = 128; s > 0; s >>= 1) {
    if (tid < s) sm[tid] += sm[tid + s];
    __syncthreads();
  }
  float r = sm[0];
  __syncthreads();
  return r;
}

__global__ __launch_bounds__(256)
void token_loss_kernel(const float* __restrict__ pmax,
                       const float* __restrict__ psum,
                       const float* __restrict__ tokl,
                       const unsigned char* __restrict__ amask,
                       const float* __restrict__ adv,
                       const float* __restrict__ oldlp,
                       float* __restrict__ acc)
{
  __shared__ float sm[256];
  const int r = blockIdx.x * 256 + threadIdx.x;  // token index, < M_
  const int b = r >> 10;                          // T_ = 1024 tokens/batch

  float gmax = -__builtin_inff();
#pragma unroll
  for (int s = 0; s < NSLOTS; ++s) gmax = fmaxf(gmax, pmax[(size_t)r * NSLOTS + s]);
  float ssum = 0.f;
#pragma unroll
  for (int s = 0; s < NSLOTS; ++s)
    ssum += psum[(size_t)r * NSLOTS + s] * __expf(pmax[(size_t)r * NSLOTS + s] - gmax);
  const float lse = gmax + __logf(ssum);
  const float lp  = tokl[r] - lse;

  const bool mk  = amask[r] != 0;
  const float pt = mk ? lp : 1.0f;               // INVALID_LOGPROB
  const float diff = pt - oldlp[r];
  const float c1 = __expf(diff);
  const float c2 = fminf(fmaxf(c1, 0.8f), 1.2f); // clip [1-eps_low, 1+eps_high]
  const float a  = adv[b];
  const float lt = -fminf(c1 * a, c2 * a);
  const float mf = mk ? 1.f : 0.f;
  const float clipf =
      (((c1 < 0.8f) && (a < 0.f)) || ((c1 > 1.2f) && (a > 0.f))) ? 1.f : 0.f;

  const float s0 = block_sum256(lt * mf, sm);            // masked loss sum
  const float s1 = block_sum256(mf, sm);                 // mask count
  const float s2 = block_sum256(clipf * mf, sm);         // clipped count
  const float s3 = block_sum256(c1, sm);                 // coef sum
  const float s4 = block_sum256(0.5f * diff * diff, sm); // kl sum

  if (threadIdx.x == 0) {
    atomicAdd(acc + b, s0);
    atomicAdd(acc + B_ + b, s1);
    atomicAdd(acc + 2 * B_ + 0, s2);
    atomicAdd(acc + 2 * B_ + 1, s3);
    atomicAdd(acc + 2 * B_ + 2, s4);
  }
}

// ---------------------------------------------------------------------------
// Finalize scalars -> d_out = [loss, clip_frac, loss, mean_coef, mean_kl].
// ---------------------------------------------------------------------------
__global__ void finalize_kernel(const float* __restrict__ acc, float* __restrict__ out) {
  if (threadIdx.x == 0 && blockIdx.x == 0) {
    float loss = 0.f, total = 0.f;
    for (int b = 0; b < B_; ++b) {
      loss  += acc[b] / fmaxf(acc[B_ + b], 1.f);
      total += acc[B_ + b];
    }
    loss /= (float)B_;
    const float clip_frac = acc[2 * B_ + 0] / fmaxf(total, 1.f);
    const float mean_coef = acc[2 * B_ + 1] / (float)(B_ * T_) / (float)B_;
    const float mean_kl   = acc[2 * B_ + 2] / (float)(B_ * T_) / (float)B_;
    out[0] = loss;
    out[1] = clip_frac;
    out[2] = loss;
    out[3] = mean_coef;
    out[4] = mean_kl;
  }
}

__global__ void init_acc(float* __restrict__ acc) {
  if (threadIdx.x < 2 * B_ + 3) acc[threadIdx.x] = 0.f;
}

// ---------------------------------------------------------------------------
extern "C" void kernel_launch(void* const* d_in, const int* in_sizes, int n_in,
                              void* d_out, int out_size, void* d_ws, size_t ws_size,
                              hipStream_t stream) {
  (void)in_sizes; (void)n_in; (void)out_size; (void)ws_size;

  const float* X      = (const float*)d_in[0];  // (B*T, H) fp32
  const float* W      = (const float*)d_in[1];  // (V, H)   fp32
  const float* bias   = (const float*)d_in[2];  // (V,)     fp32
  const int*   tok    = (const int*)d_in[3];    // (B, T)   int32
  const unsigned char* amask = (const unsigned char*)d_in[4];  // (B, T) bool
  const float* adv    = (const float*)d_in[5];  // (B,)     fp32
  const float* oldlp  = (const float*)d_in[6];  // (B, T)   fp32
  float* out = (float*)d_out;

  char* ws = (char*)d_ws;
  __bf16* Whi = (__bf16*)ws; ws += (size_t)V_ * H_ * sizeof(__bf16);
  __bf16* Wlo = (__bf16*)ws; ws += (size_t)V_ * H_ * sizeof(__bf16);
  __bf16* Xhi = (__bf16*)ws; ws += (size_t)M_ * H_ * sizeof(__bf16);
  __bf16* Xlo = (__bf16*)ws; ws += (size_t)M_ * H_ * sizeof(__bf16);
  float* pmax = (float*)ws;  ws += (size_t)M_ * NSLOTS * sizeof(float);
  float* psum = (float*)ws;  ws += (size_t)M_ * NSLOTS * sizeof(float);
  float* tokl = (float*)ws;  ws += (size_t)M_ * sizeof(float);
  float* acc  = (float*)ws;

  const int w4 = (V_ * H_) / 4;   // 16,384,000 float4 groups
  const int x4 = (M_ * H_) / 4;   //  2,097,152 float4 groups

  init_acc<<<dim3(1), dim3(32), 0, stream>>>(acc);
  split_bf16<<<dim3((w4 + 255) / 256), dim3(256), 0, stream>>>(W, Whi, Wlo, w4);
  split_bf16<<<dim3((x4 + 255) / 256), dim3(256), 0, stream>>>(X, Xhi, Xlo, x4);
  grpo_lse_gemm<<<dim3(M_ / 64, NCHUNK), dim3(256), 0, stream>>>(
      Xhi, Xlo, Whi, Wlo, bias, pmax, psum);
  tok_logit_kernel<<<dim3(M_ / 8), dim3(256), 0, stream>>>(X, W, bias, tok, tokl);
  token_loss_kernel<<<dim3(M_ / 256), dim3(256), 0, stream>>>(
      pmax, psum, tokl, amask, adv, oldlp, acc);
  finalize_kernel<<<dim3(1), dim3(1), 0, stream>>>(acc, out);
}
